// lightGCN_LQTransformer_83846351552979
// MI455X (gfx1250) — compile-verified, matrix operand order checked by reading, not run on hardware
//
#include <hip/hip_runtime.h>
#include <math.h>

// ---------------------------------------------------------------------------
// MI455X (gfx1250) implementation of lightGCN_LQTransformer forward.
// wave32, WMMA bf16 (v_wmma_f32_16x16x32_bf16) for all GEMMs, fp32 state.
// Weights repacked into WMMA-operand order so each 32x16 B tile loads as two
// coalesced b128 vector loads per lane (no scalar u16 assembly).
// ---------------------------------------------------------------------------

#define B_  128
#define S_  512
#define H_  384
#define E_  128
#define HALF_ 192
#define FOURH_ 1536
#define NTOK (B_ * S_)          // 65536
#define ACAT_K 1280             // fused gather width

typedef __bf16 bf16_t;
typedef __attribute__((ext_vector_type(16))) __bf16 v16bf;
typedef __attribute__((ext_vector_type(8)))  __bf16 v8bf;
typedef __attribute__((ext_vector_type(8)))  float  v8f;

union BFU { unsigned short u; bf16_t b; };

static __device__ __forceinline__ bf16_t f2bf(float f) {
    unsigned u = __float_as_uint(f);
    u += 0x7FFFu + ((u >> 16) & 1u);          // round-to-nearest-even
    BFU x; x.u = (unsigned short)(u >> 16);
    return x.b;
}
static __device__ __forceinline__ float bf2f(bf16_t h) {
    BFU x; x.b = h;
    return __uint_as_float(((unsigned)x.u) << 16);
}

// ----- WMMA tile loaders (ISA 7.12.2 layouts, wave32) -----------------------
// A 16x32 bf16: lane L<16 -> row m0+L, K = kb..kb+7 and kb+16..kb+23 (kb=k0);
//               lane L>=16 -> same row, kb = k0+8.
static __device__ __forceinline__ v16bf load_a_tile(const bf16_t* __restrict__ A,
                                                    int lda, int m0, int k0, int lane) {
    int row = m0 + (lane & 15);
    int kb  = k0 + ((lane >> 4) << 3);
    const v8bf lo = *(const v8bf*)(A + (size_t)row * lda + kb);
    const v8bf hi = *(const v8bf*)(A + (size_t)row * lda + kb + 16);
    v16bf r;
#pragma unroll
    for (int e = 0; e < 8; ++e) { r[e] = lo[e]; r[8 + e] = hi[e]; }
    return r;
}
// Packed B: per 32x16 tile, 512 bf16 laid out as [half][lane][8]; both halves
// are fully coalesced b128 loads. Tile order: (k0/32) major, (n0/16) minor.
static __device__ __forceinline__ v16bf load_b_packed(const bf16_t* __restrict__ Bp,
                                                      int N, int k0, int n0, int lane) {
    const bf16_t* base = Bp + (((size_t)(k0 >> 5) * (N >> 4) + (n0 >> 4)) << 9);
    const v8bf lo = *(const v8bf*)(base + lane * 8);
    const v8bf hi = *(const v8bf*)(base + 256 + lane * 8);
    v16bf r;
#pragma unroll
    for (int e = 0; e < 8; ++e) { r[e] = lo[e]; r[8 + e] = hi[e]; }
    return r;
}

// ---------------------------------------------------------------------------
// Generic bf16 GEMM: C = A(MxK) * Bpacked(KxN) + bias, opt relu + bf16 mirror.
// 4 waves per block; each wave computes NT 16x16 tiles along N (A reuse).
// ---------------------------------------------------------------------------
template <int NT>
__global__ void gemm_bf16_wmma(const bf16_t* __restrict__ A,
                               const bf16_t* __restrict__ Bp,
                               const float*  __restrict__ bias,
                               float*        __restrict__ C,
                               bf16_t*       __restrict__ Cbf,
                               int M, int N, int K, int relu) {
    const int lane = threadIdx.x & 31;
    const int wave = threadIdx.x >> 5;
    const int m0    = blockIdx.x * 16;
    const int nbase = blockIdx.y * (4 * 16 * NT) + wave * 16 * NT;
    if (m0 >= M || nbase >= N) return;   // wave-uniform: EXEC stays all-1s

    v8f acc[NT];
#pragma unroll
    for (int j = 0; j < NT; ++j)
#pragma unroll
        for (int e = 0; e < 8; ++e) acc[j][e] = 0.0f;

#pragma unroll 2
    for (int k0 = 0; k0 < K; k0 += 32) {
        if (k0 + 32 < K)    // streamed operand: pull next K-chunk toward L0/L2
            __builtin_prefetch(A + (size_t)(m0 + (lane & 15)) * K + k0 + 32, 0, 3);
        v16bf a = load_a_tile(A, K, m0, k0, lane);
#pragma unroll
        for (int j = 0; j < NT; ++j) {
            v16bf b = load_b_packed(Bp, N, k0, nbase + j * 16, lane);
            acc[j] = __builtin_amdgcn_wmma_f32_16x16x32_bf16(
                         false, a, false, b, (short)0, acc[j], false, false);
        }
    }
    const int rbase = m0 + ((lane >> 4) << 3);
    const int cl    = lane & 15;
#pragma unroll
    for (int j = 0; j < NT; ++j) {
        int n0 = nbase + j * 16;
#pragma unroll
        for (int r = 0; r < 8; ++r) {
            int row = rbase + r, col = n0 + cl;
            float v = acc[j][r] + (bias ? bias[col] : 0.0f);
            if (relu) v = fmaxf(v, 0.0f);
            C[(size_t)row * N + col] = v;
            if (Cbf) Cbf[(size_t)row * N + col] = f2bf(v);
        }
    }
}

// ---------------------------------------------------------------------------
// Weight prep kernels
// ---------------------------------------------------------------------------
// Pack fp32 matrix into WMMA B-operand order (bf16).
// trans==0: val(k,n) = src[k*N + n]   (src stored K x N)
// trans==1: val(k,n) = src[n*K + k]   (src stored N x K, i.e. used as W^T)
__global__ void k_pack_b(bf16_t* __restrict__ dst, const float* __restrict__ src,
                         int K, int N, int trans) {
    int id = blockIdx.x * 256 + threadIdx.x;
    if (id >= K * N) return;
    int e    = id & 7;
    int lane = (id >> 3) & 31;
    int h    = (id >> 8) & 1;
    int tile = id >> 9;
    int ntn  = N >> 4;
    int kt = tile / ntn, nt = tile % ntn;
    int col = (nt << 4) + (lane & 15);
    int k   = (kt << 5) + ((lane >> 4) << 3) + h * 16 + e;
    float v = trans ? src[(size_t)col * K + k] : src[(size_t)k * N + col];
    dst[id] = f2bf(v);
}
// Fused W_cat' (1280 x 192) fp32: LGCN blocks folded in (W_lgcn @ W_cat_block).
__global__ void k_build_wcat(float* dst, const float* Wcat,
                             const float* Wlt, const float* Wlq) {
    int id = blockIdx.x * 256 + threadIdx.x;
    if (id >= ACAT_K * HALF_) return;
    int r = id / HALF_, c = id % HALF_;
    float v;
    if (r < 256) {                         // interaction + test emb blocks
        v = Wcat[(size_t)r * HALF_ + c];
    } else if (r < 512) {                  // test LGCN: Wlt(256x128) @ Wcat[256:384]
        int rr = r - 256; float s = 0.f;
        for (int k = 0; k < 128; ++k)
            s += Wlt[(size_t)rr * 128 + k] * Wcat[(size_t)(256 + k) * HALF_ + c];
        v = s;
    } else if (r < 640) {                  // question emb (rows 384..511)
        v = Wcat[(size_t)(r - 512 + 384) * HALF_ + c];
    } else if (r < 768) {                  // question LGCN: Wlq(128x128) @ Wcat[512:640]
        int rr = r - 640; float s = 0.f;
        for (int k = 0; k < 128; ++k)
            s += Wlq[(size_t)rr * 128 + k] * Wcat[(size_t)(512 + k) * HALF_ + c];
        v = s;
    } else {                               // tag/big/month/dayname (rows 640..1151)
        v = Wcat[(size_t)(r - 768 + 640) * HALF_ + c];
    }
    dst[id] = v;
}
// Fused bias' = b_cat + b_lgcn_t @ Wcat[256:384] + b_lgcn_q @ Wcat[512:640]
__global__ void k_build_biasp(float* dst, const float* bcat, const float* Wcat,
                              const float* blt, const float* blq) {
    int c = threadIdx.x;
    if (c >= HALF_) return;
    float s = bcat[c];
    for (int k = 0; k < 128; ++k) s += blt[k] * Wcat[(size_t)(256 + k) * HALF_ + c];
    for (int k = 0; k < 128; ++k) s += blq[k] * Wcat[(size_t)(512 + k) * HALF_ + c];
    dst[c] = s;
}
__global__ void k_bias_g(float* dst, const float* bih, const float* bhh) {
    int j = blockIdx.x * 256 + threadIdx.x;
    if (j < FOURH_) dst[j] = bih[j] + bhh[j];
}
__global__ void k_init_state(bf16_t* h0, float* c) {
    int id = blockIdx.x * 256 + threadIdx.x;
    if (id < B_ * H_) { h0[id] = f2bf(0.0f); c[id] = 0.0f; }
}

// ---------------------------------------------------------------------------
// Gather fused categorical matrix A' (NTOK x 1280) bf16, (b,s) row order.
// ---------------------------------------------------------------------------
__global__ void k_gather(bf16_t* __restrict__ Ap,
                         const int* test, const int* question, const int* tag,
                         const int* interaction, const int* big, const int* month,
                         const int* dayname,
                         const float* e_int, const float* e_test, const float* e_q,
                         const float* e_tag, const float* e_big, const float* e_mo,
                         const float* e_day, const float* Tmat, const float* Qmat) {
    int t = blockIdx.x;
    int it = interaction[t], te = test[t], qi = question[t], tg = tag[t];
    int bg = big[t], mo = month[t], dy = dayname[t];
    bf16_t* row = Ap + (size_t)t * ACAT_K;
    for (int e = threadIdx.x; e < ACAT_K; e += 256) {
        float v;
        if      (e <  128) v = e_int [(size_t)it * 128 + e];
        else if (e <  256) v = e_test[(size_t)te * 128 + (e - 128)];
        else if (e <  512) v = Tmat  [(size_t)(6999 + te) * 256 + (e - 256)];
        else if (e <  640) v = e_q   [(size_t)qi * 128 + (e - 512)];
        else if (e <  768) v = Qmat  [(size_t)(6999 + qi) * 128 + (e - 640)];
        else if (e <  896) v = e_tag [(size_t)tg * 128 + (e - 768)];
        else if (e < 1024) v = e_big [(size_t)bg * 128 + (e - 896)];
        else if (e < 1152) v = e_mo  [(size_t)mo * 128 + (e - 1024)];
        else               v = e_day [(size_t)dy * 128 + (e - 1152)];
        row[e] = f2bf(v);
    }
}

// ---------------------------------------------------------------------------
// Build X: numeric branch (7->192 + LN) ++ embed_cat, +pos, LN1.
// Input token order (b,s); output X in (s,b) order, fp32 + bf16.
// One block (384 threads) per token.
// ---------------------------------------------------------------------------
__global__ void k_build_x(const float* __restrict__ ecat,
                          const float* n0, const float* n1, const float* n2,
                          const float* n3, const float* n4, const float* n5,
                          const float* n6,
                          const float* Wnum, const float* bnum,
                          const float* gnum, const float* bnln,
                          const float* pos, const float* g1, const float* b1,
                          float* __restrict__ X, bf16_t* __restrict__ Xbf) {
    __shared__ float shn[HALF_];
    __shared__ float rA[512], rB[512];
    int t = blockIdx.x, j = threadIdx.x;
    int b = t / S_, s = t % S_;

    float catv = 0.f;
    if (j < HALF_) {
        catv = ecat[(size_t)t * HALF_ + j];
    } else {
        int jn = j - HALF_;
        float nf[7] = { n0[t], n1[t], n2[t], n3[t], n4[t], n5[t], n6[t] };
        float d = bnum[jn];
#pragma unroll
        for (int f = 0; f < 7; ++f) d += nf[f] * Wnum[f * HALF_ + jn];
        shn[jn] = d;
    }
    __syncthreads();
    if (j < 256) {
        float v = (j < HALF_) ? shn[j] : 0.f;
        rA[j] = v; rB[j] = v * v;
    }
    __syncthreads();
    for (int st = 128; st > 0; st >>= 1) {
        if (j < st) { rA[j] += rA[j + st]; rB[j] += rB[j + st]; }
        __syncthreads();
    }
    float mn = rA[0] / (float)HALF_;
    float vr = rB[0] / (float)HALF_ - mn * mn;
    float xnum = 0.f;
    if (j >= HALF_) {
        int jn = j - HALF_;
        xnum = (shn[jn] - mn) * rsqrtf(vr + 1e-5f) * gnum[jn] + bnln[jn];
    }
    __syncthreads();

    float xf = ((j < HALF_) ? catv : xnum) + pos[(size_t)s * H_ + j];
    rA[j] = xf; rB[j] = xf * xf;
    if (j < 128) { rA[j + 384] = 0.f; rB[j + 384] = 0.f; }
    __syncthreads();
    for (int st = 256; st > 0; st >>= 1) {
        if (j < st) { rA[j] += rA[j + st]; rB[j] += rB[j + st]; }
        __syncthreads();
    }
    float m2 = rA[0] / (float)H_;
    float v2 = rB[0] / (float)H_ - m2 * m2;
    float y = (xf - m2) * rsqrtf(v2 + 1e-5f) * g1[j] + b1[j];
    size_t o = ((size_t)s * B_ + b) * H_ + j;
    X[o] = y; Xbf[o] = f2bf(y);
}

// ---------------------------------------------------------------------------
// Attention for one batch row: scores (512 dots) + softmax + ctx reduction.
// ---------------------------------------------------------------------------
__global__ void k_attn(const float* __restrict__ q, const float* __restrict__ Kb,
                       const float* __restrict__ Vb,
                       float* __restrict__ ctx, bf16_t* __restrict__ ctxbf) {
    __shared__ float qs[H_];
    __shared__ float sc[S_];
    __shared__ float red[256];
    int b = blockIdx.x, t = threadIdx.x;
    for (int h = t; h < H_; h += 256) qs[h] = q[(size_t)b * H_ + h];
    __syncthreads();
    for (int s = t; s < S_; s += 256) {
        const float* kr = Kb + ((size_t)s * B_ + b) * H_;
        float d = 0.f;
        for (int h = 0; h < H_; ++h) d += qs[h] * kr[h];
        sc[s] = d * 0.05103103630798287f;     // 1/sqrt(384)
    }
    __syncthreads();
    float m = -3.4e38f;
    for (int s = t; s < S_; s += 256) m = fmaxf(m, sc[s]);
    red[t] = m; __syncthreads();
    for (int st = 128; st > 0; st >>= 1) {
        if (t < st) red[t] = fmaxf(red[t], red[t + st]);
        __syncthreads();
    }
    m = red[0]; __syncthreads();
    float sum = 0.f;
    for (int s = t; s < S_; s += 256) { float e = __expf(sc[s] - m); sc[s] = e; sum += e; }
    red[t] = sum; __syncthreads();
    for (int st = 128; st > 0; st >>= 1) {
        if (t < st) red[t] += red[t + st];
        __syncthreads();
    }
    float inv = 1.0f / red[0];
    __syncthreads();
    float a0 = 0.f, a1 = 0.f;
    int h0 = t, h1 = t + 256;
    for (int s = 0; s < S_; ++s) {
        const float* vr = Vb + ((size_t)s * B_ + b) * H_;
        float a = sc[s];
        a0 += a * vr[h0];
        if (h1 < H_) a1 += a * vr[h1];
    }
    size_t o0 = (size_t)b * H_ + h0;
    ctx[o0] = a0 * inv; ctxbf[o0] = f2bf(a0 * inv);
    if (h1 < H_) {
        size_t o1 = (size_t)b * H_ + h1;
        ctx[o1] = a1 * inv; ctxbf[o1] = f2bf(a1 * inv);
    }
}

// X2 = X + broadcast(ctxo over s); refresh bf16 mirror.
__global__ void k_add_ctx(float* __restrict__ X, bf16_t* __restrict__ Xbf,
                          const float* __restrict__ ctxo) {
    size_t id = (size_t)blockIdx.x * 256 + threadIdx.x;
    if (id >= (size_t)NTOK * H_) return;
    int h = (int)(id % H_);
    int b = (int)((id / H_) & (B_ - 1));       // token order (s,b), B=128
    float v = X[id] + ctxo[(size_t)b * H_ + h];
    X[id] = v; Xbf[id] = f2bf(v);
}

// ---------------------------------------------------------------------------
// One LSTM step. Block = 4 waves; wave w computes gate chunk w (i/f/g/o) of a
// 16x16 (batch x hidden) patch via 12 WMMA ops; recombine in LDS; ping-pong h.
// ---------------------------------------------------------------------------
__global__ void k_lstm_step(const float* __restrict__ Gs,          // B x 1536
                            const bf16_t* __restrict__ hin,        // B x 384
                            const bf16_t* __restrict__ WhhP,       // packed 384x1536
                            float* __restrict__ c,                 // B x 384
                            bf16_t* __restrict__ hout) {           // B x 384
    __shared__ float lg[4][16][16];
    const int lane = threadIdx.x & 31;
    const int wave = threadIdx.x >> 5;
    const int m0 = blockIdx.x * 16;            // batch tile
    const int jt = blockIdx.y * 16;            // hidden-col tile in [0,384)
    const int n0 = wave * H_ + jt;             // gate-chunk column in [0,1536)

    v8f acc;
#pragma unroll
    for (int e = 0; e < 8; ++e) acc[e] = 0.0f;
#pragma unroll
    for (int k0 = 0; k0 < H_; k0 += 32) {
        v16bf a = load_a_tile(hin, H_, m0, k0, lane);
        v16bf b = load_b_packed(WhhP, FOURH_, k0, n0, lane);
        acc = __builtin_amdgcn_wmma_f32_16x16x32_bf16(
                  false, a, false, b, (short)0, acc, false, false);
    }
    const int rbase = (lane >> 4) << 3;
    const int cl = lane & 15;
#pragma unroll
    for (int r = 0; r < 8; ++r) {
        int lrow = rbase + r;
        lg[wave][lrow][cl] =
            acc[r] + Gs[(size_t)(m0 + lrow) * FOURH_ + n0 + cl];
    }
    __syncthreads();
    for (int cell = threadIdx.x; cell < 256; cell += 128) {
        int bi = cell >> 4, ji = cell & 15;
        float iv = lg[0][bi][ji], fv = lg[1][bi][ji];
        float gv = lg[2][bi][ji], ov = lg[3][bi][ji];
        size_t idx = (size_t)(m0 + bi) * H_ + jt + ji;
        float ig = 1.f / (1.f + __expf(-iv));
        float fg = 1.f / (1.f + __expf(-fv));
        float og = 1.f / (1.f + __expf(-ov));
        float cn = fg * c[idx] + ig * tanhf(gv);
        c[idx] = cn;
        hout[idx] = f2bf(og * tanhf(cn));
    }
}

// LN over 384 per row (used for LN2 on final h). Block = 384 threads.
__global__ void k_ln384(const bf16_t* __restrict__ hin, const float* g,
                        const float* bb, float* __restrict__ out,
                        bf16_t* __restrict__ outbf) {
    __shared__ float rA[512], rB[512];
    int b = blockIdx.x, j = threadIdx.x;
    float x = bf2f(hin[(size_t)b * H_ + j]);
    rA[j] = x; rB[j] = x * x;
    if (j < 128) { rA[j + 384] = 0.f; rB[j + 384] = 0.f; }
    __syncthreads();
    for (int st = 256; st > 0; st >>= 1) {
        if (j < st) { rA[j] += rA[j + st]; rB[j] += rB[j + st]; }
        __syncthreads();
    }
    float m = rA[0] / (float)H_;
    float v = rB[0] / (float)H_ - m * m;
    float y = (x - m) * rsqrtf(v + 1e-5f) * g[j] + bb[j];
    size_t o = (size_t)b * H_ + j;
    out[o] = y; outbf[o] = f2bf(y);
}

// out[b] = dot(F2[b]+skip[b], Wout) + b_out
__global__ void k_final(const float* __restrict__ F2, const float* __restrict__ skip,
                        const float* __restrict__ Wout, const float* __restrict__ bout,
                        float* __restrict__ out) {
    __shared__ float rA[512];
    int b = blockIdx.x, j = threadIdx.x;
    rA[j] = (F2[(size_t)b * H_ + j] + skip[(size_t)b * H_ + j]) * Wout[j];
    if (j < 128) rA[j + 384] = 0.f;
    __syncthreads();
    for (int st = 256; st > 0; st >>= 1) {
        if (j < st) rA[j] += rA[j + st];
        __syncthreads();
    }
    if (j == 0) out[b] = rA[0] + bout[0];
}

// ---------------------------------------------------------------------------
static void launch_gemm(const bf16_t* A, const bf16_t* Bp, const float* bias,
                        float* C, bf16_t* Cbf, int M, int N, int K, int relu,
                        hipStream_t st) {
    dim3 blk(128);
    if (N % 512 == 0) {
        dim3 g(M / 16, N / 512);
        gemm_bf16_wmma<8><<<g, blk, 0, st>>>(A, Bp, bias, C, Cbf, M, N, K, relu);
    } else if (N % 192 == 0) {
        dim3 g(M / 16, N / 192);
        gemm_bf16_wmma<3><<<g, blk, 0, st>>>(A, Bp, bias, C, Cbf, M, N, K, relu);
    } else {
        dim3 g(M / 16, (N + 63) / 64);
        gemm_bf16_wmma<1><<<g, blk, 0, st>>>(A, Bp, bias, C, Cbf, M, N, K, relu);
    }
}

extern "C" void kernel_launch(void* const* d_in, const int* in_sizes, int n_in,
                              void* d_out, int out_size, void* d_ws, size_t ws_size,
                              hipStream_t stream) {
    (void)in_sizes; (void)n_in; (void)out_size; (void)ws_size;
    // ---- inputs ----
    const int* i_test = (const int*)d_in[0];
    const int* i_q    = (const int*)d_in[1];
    const int* i_tag  = (const int*)d_in[2];
    const int* i_int  = (const int*)d_in[3];
    const int* i_big  = (const int*)d_in[4];
    const int* i_mo   = (const int*)d_in[5];
    const int* i_day  = (const int*)d_in[6];
    const float* f_num[7];
    for (int k = 0; k < 7; ++k) f_num[k] = (const float*)d_in[7 + k];
    const float* e_int  = (const float*)d_in[14];
    const float* e_test = (const float*)d_in[15];
    const float* e_q    = (const float*)d_in[16];
    const float* e_tag  = (const float*)d_in[17];
    const float* e_big  = (const float*)d_in[18];
    const float* e_mo   = (const float*)d_in[19];
    const float* e_day  = (const float*)d_in[20];
    const float* Tmat   = (const float*)d_in[21];
    const float* Qmat   = (const float*)d_in[22];
    const float* Wlt    = (const float*)d_in[23];
    const float* blt    = (const float*)d_in[24];
    const float* Wlq    = (const float*)d_in[25];
    const float* blq    = (const float*)d_in[26];
    const float* Wcat   = (const float*)d_in[27];
    const float* bcat   = (const float*)d_in[28];
    const float* Wnum   = (const float*)d_in[29];
    const float* bnum   = (const float*)d_in[30];
    const float* gnum   = (const float*)d_in[31];
    const float* bnln   = (const float*)d_in[32];
    const float* pos    = (const float*)d_in[33];
    const float* Wqkv   = (const float*)d_in[34];
    const float* bqkv   = (const float*)d_in[35];
    const float* Wo     = (const float*)d_in[36];
    const float* bo     = (const float*)d_in[37];
    const float* g1     = (const float*)d_in[38];
    const float* b1     = (const float*)d_in[39];
    const float* Wih    = (const float*)d_in[40];
    const float* Whh    = (const float*)d_in[41];
    const float* bih    = (const float*)d_in[42];
    const float* bhh    = (const float*)d_in[43];
    const float* g2     = (const float*)d_in[44];
    const float* b2     = (const float*)d_in[45];
    const float* Wf1    = (const float*)d_in[46];
    const float* bf1    = (const float*)d_in[47];
    const float* Wf2    = (const float*)d_in[48];
    const float* bf2    = (const float*)d_in[49];
    const float* Wout   = (const float*)d_in[50];
    const float* bout   = (const float*)d_in[51];
    float* out = (float*)d_out;

    // ---- workspace layout ----
    char* p = (char*)d_ws;
    size_t off = 0;
    auto alloc = [&](size_t bytes) -> void* {
        void* r = p + off; off += (bytes + 255) & ~(size_t)255; return r;
    };
    float*  WcatF32 = (float*) alloc((size_t)ACAT_K * HALF_ * 4);
    bf16_t* WcatP  = (bf16_t*)alloc((size_t)ACAT_K * HALF_ * 2);
    float*  biasP  = (float*) alloc(HALF_ * 4);
    bf16_t* Wq_p   = (bf16_t*)alloc((size_t)H_ * H_ * 2);
    bf16_t* Wk_p   = (bf16_t*)alloc((size_t)H_ * H_ * 2);
    bf16_t* Wv_p   = (bf16_t*)alloc((size_t)H_ * H_ * 2);
    bf16_t* Wo_p   = (bf16_t*)alloc((size_t)H_ * H_ * 2);
    bf16_t* Wih_p  = (bf16_t*)alloc((size_t)H_ * FOURH_ * 2);
    bf16_t* Whh_p  = (bf16_t*)alloc((size_t)H_ * FOURH_ * 2);
    bf16_t* Wf1_p  = (bf16_t*)alloc((size_t)H_ * FOURH_ * 2);
    bf16_t* Wf2_p  = (bf16_t*)alloc((size_t)FOURH_ * H_ * 2);
    float*  biasG  = (float*) alloc(FOURH_ * 4);
    bf16_t* h0     = (bf16_t*)alloc((size_t)B_ * H_ * 2);
    bf16_t* h1     = (bf16_t*)alloc((size_t)B_ * H_ * 2);
    float*  cst    = (float*) alloc((size_t)B_ * H_ * 4);
    float*  qbuf   = (float*) alloc((size_t)B_ * H_ * 4);
    float*  ctxf   = (float*) alloc((size_t)B_ * H_ * 4);
    bf16_t* ctxb   = (bf16_t*)alloc((size_t)B_ * H_ * 2);
    float*  ctxo   = (float*) alloc((size_t)B_ * H_ * 4);
    float*  xb     = (float*) alloc((size_t)B_ * H_ * 4);
    bf16_t* xbb    = (bf16_t*)alloc((size_t)B_ * H_ * 2);
    float*  F1f    = (float*) alloc((size_t)B_ * FOURH_ * 4);
    bf16_t* F1b    = (bf16_t*)alloc((size_t)B_ * FOURH_ * 2);
    float*  F2f    = (float*) alloc((size_t)B_ * H_ * 4);
    float*  X      = (float*) alloc((size_t)NTOK * H_ * 4);
    bf16_t* Xbf    = (bf16_t*)alloc((size_t)NTOK * H_ * 2);
    // Big region: [A' | embed_cat | K | V] lives before the LSTM; the gate
    // pre-activation matrix G (402 MB) reuses it afterwards.
    char*   big    = (char*)alloc(0);
    bf16_t* Acat   = (bf16_t*)big;                                     // 168 MB
    float*  ecat   = (float*)(big + (size_t)NTOK * ACAT_K * 2);        //  50 MB
    float*  Kb     = (float*)((char*)ecat + (size_t)NTOK * HALF_ * 4); // 100 MB
    float*  Vb     = (float*)((char*)Kb   + (size_t)NTOK * H_ * 4);    // 100 MB
    float*  G      = (float*)big;                                      // 402 MB

    // ---- 1. weight prep: pack everything into WMMA-operand bf16 panels ----
    int ne = H_ * H_;
    int gp = (ne + 255) / 256;
    k_pack_b<<<gp, 256, 0, stream>>>(Wq_p, Wqkv,          H_, H_, 1);
    k_pack_b<<<gp, 256, 0, stream>>>(Wk_p, Wqkv + ne,     H_, H_, 1);
    k_pack_b<<<gp, 256, 0, stream>>>(Wv_p, Wqkv + 2 * ne, H_, H_, 1);
    k_pack_b<<<gp, 256, 0, stream>>>(Wo_p, Wo,            H_, H_, 1);
    int nih = FOURH_ * H_;
    int gih = (nih + 255) / 256;
    k_pack_b<<<gih, 256, 0, stream>>>(Wih_p, Wih, H_, FOURH_, 1);
    k_pack_b<<<gih, 256, 0, stream>>>(Whh_p, Whh, H_, FOURH_, 1);
    k_pack_b<<<gih, 256, 0, stream>>>(Wf1_p, Wf1, H_, FOURH_, 0);
    k_pack_b<<<gih, 256, 0, stream>>>(Wf2_p, Wf2, FOURH_, H_, 0);
    int ncat = ACAT_K * HALF_;
    k_build_wcat<<<(ncat + 255) / 256, 256, 0, stream>>>(WcatF32, Wcat, Wlt, Wlq);
    k_pack_b<<<(ncat + 255) / 256, 256, 0, stream>>>(WcatP, WcatF32, ACAT_K, HALF_, 0);
    k_build_biasp<<<1, 192, 0, stream>>>(biasP, bcat, Wcat, blt, blq);
    k_bias_g<<<(FOURH_ + 255) / 256, 256, 0, stream>>>(biasG, bih, bhh);
    k_init_state<<<(B_ * H_ + 255) / 256, 256, 0, stream>>>(h0, cst);

    // ---- 2. categorical path: gather + fused GEMM (65536x1280x192) ----
    k_gather<<<NTOK, 256, 0, stream>>>(Acat, i_test, i_q, i_tag, i_int, i_big,
                                       i_mo, i_day, e_int, e_test, e_q, e_tag,
                                       e_big, e_mo, e_day, Tmat, Qmat);
    launch_gemm(Acat, WcatP, biasP, ecat, nullptr, NTOK, HALF_, ACAT_K, 0, stream);

    // ---- 3. X = LN1(concat(cat, LN(num)) + pos), (s,b) order ----
    k_build_x<<<NTOK, H_, 0, stream>>>(ecat, f_num[0], f_num[1], f_num[2],
                                       f_num[3], f_num[4], f_num[5], f_num[6],
                                       Wnum, bnum, gnum, bnln, pos, g1, b1, X, Xbf);

    // ---- 4. attention ----
    launch_gemm(Xbf, Wk_p, bqkv + H_,     Kb,   nullptr, NTOK, H_, H_, 0, stream);
    launch_gemm(Xbf, Wv_p, bqkv + 2 * H_, Vb,   nullptr, NTOK, H_, H_, 0, stream);
    launch_gemm(Xbf + (size_t)(S_ - 1) * B_ * H_, Wq_p, bqkv,
                qbuf, nullptr, B_, H_, H_, 0, stream);
    k_attn<<<B_, 256, 0, stream>>>(qbuf, Kb, Vb, ctxf, ctxb);
    launch_gemm(ctxb, Wo_p, bo, ctxo, nullptr, B_, H_, H_, 0, stream);
    k_add_ctx<<<(int)(((size_t)NTOK * H_ + 255) / 256), 256, 0, stream>>>(X, Xbf, ctxo);

    // ---- 5. LSTM: precompute input gates, then 512 fused WMMA steps ----
    launch_gemm(Xbf, Wih_p, biasG, G, nullptr, NTOK, FOURH_, H_, 0, stream);
    bf16_t* hping[2] = { h0, h1 };
    for (int s = 0; s < S_; ++s) {
        k_lstm_step<<<dim3(B_ / 16, H_ / 16), 128, 0, stream>>>(
            G + (size_t)s * B_ * FOURH_, hping[s & 1], Whh_p, cst, hping[(s + 1) & 1]);
    }
    bf16_t* hfin = hping[0];   // 512 steps end back at buffer 0

    // ---- 6. LN2 + FFN + residual + output ----
    k_ln384<<<B_, H_, 0, stream>>>(hfin, g2, b2, xb, xbb);
    launch_gemm(xbb, Wf1_p, bf1, F1f, F1b, B_, FOURH_, H_, 1, stream);
    launch_gemm(F1b, Wf2_p, bf2, F2f, nullptr, B_, H_, FOURH_, 0, stream);
    k_final<<<B_, H_, 0, stream>>>(F2f, xb, Wout, bout, out);
}
